// ConditionalAttackModule_26560077758852
// MI455X (gfx1250) — matrix operand order repaired and checked by reading, not compiled, not run on hardware
//
#include <hip/hip_runtime.h>
#include <stdint.h>

typedef __attribute__((ext_vector_type(16))) _Float16 v16h;
typedef __attribute__((ext_vector_type(8)))  _Float16 v8h;
typedef __attribute__((ext_vector_type(8)))  float    v8f;

#define HID   128          // hidden size
#define G4    512          // 4*H gate width
#define BT    16           // batch rows per workgroup (one WMMA M-tile)
#define NTHR  512          // 16 waves
#define HPAD  136          // h row stride in halves (272B: +16B/row bank skew)
#define GPAD  516          // gate row stride in floats (2064B: +16B/row skew)
#define SEQ   128
#define NHIST 126          // len(returns[:-1])
#define NSAMP 100
#define NSTEP 20

// Branch-free fast activations: v_exp_f32 + v_rcp_f32, no exec-mask churn.
__device__ __forceinline__ float sigmoidf_(float x) {
  return __builtin_amdgcn_rcpf(1.f + __expf(-x));
}
__device__ __forceinline__ float tanhf_(float x) {
  float t = __expf(-2.f * fabsf(x));                    // (0,1]
  float r = (1.f - t) * __builtin_amdgcn_rcpf(1.f + t);
  return copysignf(r, x);
}

__device__ __forceinline__ unsigned pcg_hash(unsigned x) {
  x = x * 747796405u + 2891336453u;
  unsigned w = ((x >> ((x >> 28u) + 4u)) ^ x) * 277803737u;
  return (w >> 22u) ^ w;
}

__global__ __launch_bounds__(NTHR, 1)
void lstm_iwmc_kernel(const float* __restrict__ inp,
                      const float* __restrict__ prt,
                      const float* __restrict__ W_ih,
                      const float* __restrict__ W_hh,
                      const float* __restrict__ b_lstm,
                      const float* __restrict__ w_mu,
                      const float* __restrict__ b_mu,
                      const float* __restrict__ w_ls,
                      const float* __restrict__ b_ls,
                      float* __restrict__ out, int B)
{
  __shared__ __align__(16) _Float16 h_s [BT * HPAD];  // current h, f16 A-operand
  __shared__ __align__(16) _Float16 h0_s[BT * HPAD];  // h after history (rollout reset)
  __shared__ __align__(16) float    g_s [BT * GPAD];  // raw gates h@W_hh (f32)
  __shared__ float Wih_s[G4], b_s[G4];
  __shared__ float wmu_s[HID], wls_s[HID];
  __shared__ float red_mu[BT * 32], red_ls[BT * 32];  // head partial dots
  __shared__ float x_s[BT], last_s[BT];

  const int tid   = threadIdx.x;
  const int wave  = tid >> 5;
  const int lane  = tid & 31;
  const int l15   = lane & 15;
  const int rowg0 = blockIdx.x * BT;

  for (int j = tid; j < G4;      j += NTHR) { Wih_s[j] = W_ih[j]; b_s[j] = b_lstm[j]; }
  for (int j = tid; j < HID;     j += NTHR) { wmu_s[j] = w_mu[j]; wls_s[j] = w_ls[j]; }
  for (int j = tid; j < BT*HPAD; j += NTHR) h_s[j] = (_Float16)0.f;

  // ---- Preload W_hh as per-wave f16 B fragments, kept in VGPRs for all steps.
  // B 32x16 fragment, contiguous k order used here (A matches): lanes 0-15 = col n, K=kb*32+{0..15};
  // lanes 16-31 = col n, K=kb*32+{16..31}.
  v16h Bf[2][4];
  {
    const int koff = (lane < 16) ? 0 : 16;
    #pragma unroll
    for (int tn = 0; tn < 2; ++tn) {
      const int col = wave * 32 + tn * 16 + l15;
      #pragma unroll
      for (int kb = 0; kb < 4; ++kb) {
        v16h bb;
        #pragma unroll
        for (int i = 0; i < 16; ++i)
          bb[i] = (_Float16)W_hh[(kb * 32 + koff + i) * G4 + col];
        Bf[tn][kb] = bb;
      }
    }
  }

  const float bmu = b_mu[0], bls = b_ls[0];
  const int er = wave;       // elementwise: one row per wave
  const int ec = lane * 4;   // 4 hidden columns per lane
  float c_reg[4]  = {0.f, 0.f, 0.f, 0.f};
  float c_hist[4] = {0.f, 0.f, 0.f, 0.f};
  const int aoff = (lane < 16) ? 0 : 16;  // A-fragment K-offset for this lane half

  float logw = 0.f, out_reg = 0.f;                       // wave0 lanes<16 only
  float sw = 0.f, sm1 = 0.f, sqw = 0.f, sm2 = 0.f, shm = 0.f;

  __syncthreads();

  for (int t = 0; t < NHIST + NSAMP * NSTEP; ++t) {
    const bool hist = (t < NHIST);
    int s = 0, n = 0;
    if (!hist) { int u = t - NHIST; s = u / NSTEP; n = u - s * NSTEP; }

    if (!hist && n == 0) {  // rollout reset to end-of-history state
      for (int j = tid; j < BT * HPAD; j += NTHR) h_s[j] = h0_s[j];
      #pragma unroll
      for (int q = 0; q < 4; ++q) c_reg[q] = c_hist[q];
      if (wave == 0 && lane < 16) { x_s[l15] = last_s[l15]; logw = 0.f; }
      __syncthreads();
    }

    if (hist && tid < BT) {  // x_t = returns[t] computed on the fly
      const int rg = rowg0 + tid;
      float a0 = inp[ t      * B + rg] * (1.f + prt[ t      * B + rg]);
      float a1 = inp[(t + 1) * B + rg] * (1.f + prt[(t + 1) * B + rg]);
      x_s[tid] = a1 - a0;
    }

    // ---- WMMA phase: g = h @ W_hh (16x512x128 per WG, 8 WMMA per wave) ----
    v8f acc0 = {}; v8f acc1 = {};
    #pragma unroll
    for (int kb = 0; kb < 4; ++kb) {
      // A 16x32 fragment, contiguous k order (matches Bf): lane l<16 row l K={0..15}; lane l+16 row l K={16..31}
      const _Float16* hp = &h_s[l15 * HPAD + kb * 32 + aoff];
      v8h lo = *(const v8h*)(hp);
      v8h hi = *(const v8h*)(hp + 8);
      v16h a;
      #pragma unroll
      for (int i = 0; i < 8; ++i) { a[i] = lo[i]; a[i + 8] = hi[i]; }
      acc0 = __builtin_amdgcn_wmma_f32_16x16x32_f16(false, a, false, Bf[0][kb],
                                                    (short)0, acc0, false, false);
      acc1 = __builtin_amdgcn_wmma_f32_16x16x32_f16(false, a, false, Bf[1][kb],
                                                    (short)0, acc1, false, false);
    }
    {  // C/D layout: VGPR v, lanes<16 -> M=v; lanes>=16 -> M=v+8; N=lane&15
      const int row0 = (lane < 16) ? 0 : 8;
      const int cb = wave * 32 + l15;
      #pragma unroll
      for (int v = 0; v < 8; ++v) {
        g_s[(row0 + v) * GPAD + cb     ] = acc0[v];
        g_s[(row0 + v) * GPAD + cb + 16] = acc1[v];
      }
    }
    __syncthreads();

    // ---- Elementwise: gates, c/h update, head partial dots ----
    float pm = 0.f, pl = 0.f;
    const float xr = x_s[er];
    const float* gr = &g_s[er * GPAD];
    #pragma unroll
    for (int q = 0; q < 4; ++q) {
      const int j = ec + q;
      float gi = gr[j      ] + xr * Wih_s[j      ] + b_s[j      ];
      float gf = gr[j + 128] + xr * Wih_s[j + 128] + b_s[j + 128];
      float gg = gr[j + 256] + xr * Wih_s[j + 256] + b_s[j + 256];
      float go = gr[j + 384] + xr * Wih_s[j + 384] + b_s[j + 384];
      float cn = sigmoidf_(gf) * c_reg[q] + sigmoidf_(gi) * tanhf_(gg);
      float hn = sigmoidf_(go) * tanhf_(cn);
      c_reg[q] = cn;
      h_s[er * HPAD + j] = (_Float16)hn;
      pm += hn * wmu_s[j];
      pl += hn * wls_s[j];
    }
    red_mu[er * 32 + lane] = pm;
    red_ls[er * 32 + lane] = pl;
    __syncthreads();

    if (hist) {
      if (t == NHIST - 1) {  // snapshot end-of-history state + last return
        for (int j = tid; j < BT * HPAD; j += NTHR) h0_s[j] = h_s[j];
        #pragma unroll
        for (int q = 0; q < 4; ++q) c_hist[q] = c_reg[q];
        if (tid < BT) {
          const int rg = rowg0 + tid;
          float a0 = inp[(SEQ - 2) * B + rg] * (1.f + prt[(SEQ - 2) * B + rg]);
          float a1 = inp[(SEQ - 1) * B + rg] * (1.f + prt[(SEQ - 1) * B + rg]);
          last_s[tid] = a1 - a0;
        }
        __syncthreads();
      }
      continue;  // uniform across block: no head phase in history
    }

    // ---- Head: mu/sigma, shifted-Gaussian sample, importance weight ----
    if (wave == 0) {
      const float* rp = (lane < 16) ? &red_mu[l15 * 32] : &red_ls[l15 * 32];
      float sum = 0.f;
      #pragma unroll
      for (int i = 0; i < 32; ++i) sum += rp[i];
      float other = __shfl(sum, l15 + 16, 32);  // lane r grabs log-sigma sum from r+16
      if (lane < 16) {
        float mu    = sum + bmu;
        float lsx   = fminf(fmaxf(other + bls, -5.f), 2.f);
        float sigma = __expf(lsx);
        const int rg = rowg0 + l15;
        unsigned seed = (unsigned)rg * 0x9E3779B9u ^ (unsigned)s * 1315423911u
                        ^ (unsigned)n * 0x85EBCA6Bu;
        unsigned r1 = pcg_hash(seed);
        unsigned r2 = pcg_hash(seed ^ 0x68E31DA4u);
        float u1 = (float)((r1 >> 8) + 1u) * (1.f / 16777216.f);
        float u2 = (float)( r2 >> 8      ) * (1.f / 16777216.f);
        float z  = sqrtf(-2.f * __logf(u1)) * __cosf(6.28318530718f * u2);
        float xn = mu + sigma * (z + 0.5f);            // SHIFT = 0.5
        logw -= 0.5f * z + 0.125f;                     // SHIFT*z + 0.5*SHIFT^2
        x_s[l15] = xn;
        if (n == 18) out_reg = xn;                     // STEP_PRED - 1
      }
    }
    __syncthreads();

    if (n == NSTEP - 1 && wave == 0 && lane < 16) {    // MC moment accumulation
      float w  = __expf(logw);
      float ow = out_reg * w;
      sw += w; sm1 += ow; sqw += w * w; sm2 += ow * ow; shm += out_reg * w * w;
    }
  }

  if (wave == 0 && lane < 16) {
    const int rg = rowg0 + l15;
    float mean = sm1 / sw;
    float sem  = sm2 + sqw * sm1 * sm1 / (sw * sw) - 2.f * shm * sm1 / sw;
    sem = sqrtf(sem / (float)(NSAMP * (NSAMP - 1)));
    out[rg]     = mean;
    out[B + rg] = sem;
  }
}

extern "C" void kernel_launch(void* const* d_in, const int* in_sizes, int n_in,
                              void* d_out, int out_size, void* d_ws, size_t ws_size,
                              hipStream_t stream) {
  const float* inp  = (const float*)d_in[0];
  const float* prt  = (const float*)d_in[1];
  const float* Wih  = (const float*)d_in[2];
  const float* Whh  = (const float*)d_in[3];
  const float* bl   = (const float*)d_in[4];
  const float* wmu  = (const float*)d_in[5];
  const float* bmu  = (const float*)d_in[6];
  const float* wls  = (const float*)d_in[7];
  const float* bls  = (const float*)d_in[8];
  float* out = (float*)d_out;

  const int B = in_sizes[0] / SEQ;  // 8192
  dim3 grid(B / BT), block(NTHR);
  lstm_iwmc_kernel<<<grid, block, 0, stream>>>(inp, prt, Wih, Whh, bl,
                                               wmu, bmu, wls, bls, out, B);
}
